// BiMambaLRP_17257178596142
// MI455X (gfx1250) — compile-verified, matrix operand order checked by reading, not run on hardware
//
#include <hip/hip_runtime.h>
#include <hip/hip_bf16.h>

// BiMamba (bidirectional Mamba block) for MI455X / gfx1250.
// All four projections run on v_wmma_f32_16x16x32_bf16 (bf16 operands,
// fp32 accumulate). Conv / softplus / selective-scan / SiLU gating run in
// fp32 VALU. Reverse direction handled by index remapping in GEMM1 (read)
// and GEMM4 (write, accumulating). Workspace requirement ~100 MB.
//
// GEMM: wave tile 32(M) x 64(N) = 8 WMMAs per K-step over 6 fragments;
// row/col base pointers hoisted out of the K-loop so the inner loop is
// pure global_load_b128 + v_wmma + pointer arithmetic. Prefetch is
// unconditional (speculative prefetch past the end is dropped by HW).

#define B_  2
#define L_  1024
#define D_  1024
#define DI_ 2048
#define M_  (B_ * L_)     // 2048 rows (B*L)
#define DT_RANK_ 64
#define D_STATE_ 16
#define XDBL_ 96          // DT_RANK + 2*D_STATE

typedef __attribute__((ext_vector_type(16))) __bf16 v16bf;
typedef __attribute__((ext_vector_type(8)))  __bf16 v8bf;
typedef __attribute__((ext_vector_type(8)))  float  v8f;

__device__ __forceinline__ unsigned short f32_to_bf16_rne(float f) {
    unsigned int u = __float_as_uint(f);
    u += 0x7FFFu + ((u >> 16) & 1u);   // round-to-nearest-even
    return (unsigned short)(u >> 16);
}

// ---------------------------------------------------------------------------
// fp32 -> bf16 bulk convert
// ---------------------------------------------------------------------------
__global__ __launch_bounds__(256)
void k_f32_to_bf16(const float* __restrict__ src, unsigned short* __restrict__ dst, long n) {
    long i = (long)blockIdx.x * blockDim.x + threadIdx.x;
    if (i < n) dst[i] = f32_to_bf16_rne(src[i]);
}

// ---------------------------------------------------------------------------
// Generic bf16 WMMA GEMM:  C[M,N] = A[M,K] * W[N,K]^T   (both K-contiguous)
//   revA  : read A row (b, L-1-l) instead of (b, l)   (reverse-time input)
//   revC  : write C row (b, L-1-l)                    (reverse-time output)
//   accum : C += result  (second direction's output GEMM)
//   epi=1 : softplus(val + bias[col])                 (dt projection)
//   Cb    : if non-null, also store bf16 copy of C (fused downstream convert)
// Per-wave tile 32x64 (2x4 WMMA); 8 waves (4Mx2N) -> 128x128 block tile.
// Operand fragments follow the CDNA5 16-bit A/B layout:
//   lanes 0-15 : K = [k..k+7] , [k+16..k+23]   (row/col = lane&15)
//   lanes 16-31: K = [k+8..k+15], [k+24..k+31]
// ---------------------------------------------------------------------------
__global__ __launch_bounds__(256)
void k_gemm_bf16_wmma(const __bf16* __restrict__ A, int lda,
                      const __bf16* __restrict__ W, int ldb,
                      float* __restrict__ C, int ldc,
                      int M, int N, int K, int Lseq,
                      int revA, int revC, int accum,
                      const float* __restrict__ bias, int epi,
                      unsigned short* __restrict__ Cb)
{
    const int lane  = threadIdx.x & 31;
    const int wave  = threadIdx.x >> 5;   // 0..7
    const int waveM = wave >> 1;          // 0..3
    const int waveN = wave & 1;           // 0..1
    const int half  = lane >> 4;          // 0 / 1
    const int r     = lane & 15;

    const int tileM = blockIdx.y * 128 + waveM * 32;   // 2 x 16 rows
    const int tileN = blockIdx.x * 128 + waveN * 64;   // 4 x 16 cols

    const int k0 = half * 8;
    const int k1 = 16 + half * 8;

    // Hoist row/col base pointers out of the K-loop (rev/clamp mapping is
    // K-invariant); the inner loop is then loads + WMMA only.
    const __bf16* apr[2];
    #pragma unroll
    for (int i = 0; i < 2; ++i) {
        int m = tileM + i * 16 + r;
        if (m >= M) m = M - 1;                         // clamp (store is guarded)
        if (revA) { int bb = m / Lseq, ll = m % Lseq; m = bb * Lseq + (Lseq - 1 - ll); }
        apr[i] = A + (long)m * lda;
    }
    const __bf16* bpr[4];
    #pragma unroll
    for (int j = 0; j < 4; ++j) {
        int n = tileN + j * 16 + r;
        if (n >= N) n = N - 1;
        bpr[j] = W + (long)n * ldb;
    }

    v8f acc[2][4] = {};

    for (int kk = 0; kk < K; kk += 32) {
        v16bf a[2], b[4];
        #pragma unroll
        for (int i = 0; i < 2; ++i) {
            union { v16bf v; v8bf h[2]; } t;
            t.h[0] = *(const v8bf*)(apr[i] + kk + k0);
            t.h[1] = *(const v8bf*)(apr[i] + kk + k1);
            a[i] = t.v;
        }
        #pragma unroll
        for (int j = 0; j < 4; ++j) {
            union { v16bf v; v8bf h[2]; } t;
            t.h[0] = *(const v8bf*)(bpr[j] + kk + k0);
            t.h[1] = *(const v8bf*)(bpr[j] + kk + k1);
            b[j] = t.v;
        }
        // Unconditional speculative prefetch of the next K-slice; HW drops
        // prefetches whose translation fails, so no end-of-loop guard needed.
        __builtin_prefetch(apr[0] + kk + 32, 0, 1);    // global_prefetch_b8
        __builtin_prefetch(bpr[0] + kk + 32, 0, 1);

        #pragma unroll
        for (int i = 0; i < 2; ++i)
            #pragma unroll
            for (int j = 0; j < 4; ++j)
                acc[i][j] = __builtin_amdgcn_wmma_f32_16x16x32_bf16(
                    false, a[i], false, b[j], (short)0, acc[i][j], false, false);
    }

    // C/D layout: within a 16x16 tile, row = half*8 + v, col = lane&15.
    #pragma unroll
    for (int i = 0; i < 2; ++i) {
        #pragma unroll
        for (int j = 0; j < 4; ++j) {
            int col = tileN + j * 16 + r;
            if (col >= N) continue;
            #pragma unroll
            for (int v = 0; v < 8; ++v) {
                int row = tileM + i * 16 + half * 8 + v;
                if (row >= M) continue;
                int srow = row;
                if (revC) { int bb = row / Lseq, ll = row % Lseq; srow = bb * Lseq + (Lseq - 1 - ll); }
                float val = acc[i][j][v];
                if (epi == 1) {
                    float t = val + bias[col];
                    val = (t > 20.f) ? t : log1pf(__expf(t));   // softplus
                }
                long cidx = (long)srow * ldc + col;
                if (accum) C[cidx] += val; else C[cidx] = val;
                if (Cb) Cb[cidx] = f32_to_bf16_rne(val);
            }
        }
    }
}

// ---------------------------------------------------------------------------
// Causal depthwise conv (K=4) + bias + SiLU over xz[..., 0:DI].
// Emits xc in fp32 (scan input u) and bf16 (Wx GEMM operand).
// ---------------------------------------------------------------------------
__global__ __launch_bounds__(256)
void k_conv_silu(const float* __restrict__ xz,      // (B,L,2*DI)
                 const float* __restrict__ Wc,      // (DI,4)
                 const float* __restrict__ bc,      // (DI,)
                 float* __restrict__ xc_f,          // (B,L,DI)
                 unsigned short* __restrict__ xc_b) // (B,L,DI) bf16
{
    long idx = (long)blockIdx.x * blockDim.x + threadIdx.x;
    if (idx >= (long)M_ * DI_) return;
    int d = (int)(idx % DI_);
    int l = (int)((idx / DI_) % L_);
    int b = (int)(idx / ((long)DI_ * L_));

    float acc = bc[d];
    #pragma unroll
    for (int k = 0; k < 4; ++k) {
        int l2 = l - 3 + k;
        if (l2 >= 0)
            acc += xz[((long)(b * L_ + l2)) * (2 * DI_) + d] * Wc[d * 4 + k];
    }
    float s = acc / (1.f + __expf(-acc));   // SiLU
    xc_f[idx] = s;
    xc_b[idx] = f32_to_bf16_rne(s);
}

// ---------------------------------------------------------------------------
// Selective scan. 16 lanes per (b,d) channel: lane n owns state n.
// h_n <- exp(dt*A_n)*h_n + dt*B_n*u ;  y = sum_n h_n*C_n + u*Dskip
// 4-step shfl_xor butterfly inside each 16-lane group (wave32 native).
// Output gated with silu(z), written as bf16 for the Wout WMMA GEMM.
// ---------------------------------------------------------------------------
__global__ __launch_bounds__(256)
void k_selective_scan(const float* __restrict__ u,     // (B,L,DI) xc fp32
                      const float* __restrict__ dt,    // (B,L,DI)
                      const float* __restrict__ xdbl,  // (B,L,96): [64:80]=Bm [80:96]=C
                      const float* __restrict__ Alog,  // (DI,16)
                      const float* __restrict__ Dskip, // (DI,)
                      const float* __restrict__ xz,    // (B,L,2*DI): z at [DI:]
                      unsigned short* __restrict__ yg) // (B,L,DI) bf16
{
    int ch = blockIdx.x * 16 + (threadIdx.x >> 4);      // channel (b,d)
    int n  = threadIdx.x & 15;                          // state index
    if (ch >= B_ * DI_) return;
    int b = ch / DI_;
    int d = ch % DI_;

    const float A_n = -__expf(Alog[d * D_STATE_ + n]);
    const float dsk = Dskip[d];
    float h = 0.f;

    for (int l = 0; l < L_; ++l) {
        long  base = (long)(b * L_ + l);
        float dtv  = dt[base * DI_ + d];
        float uv   = u[base * DI_ + d];
        float Bv   = xdbl[base * XDBL_ + DT_RANK_ + n];
        float Cv   = xdbl[base * XDBL_ + DT_RANK_ + D_STATE_ + n];

        h = __expf(dtv * A_n) * h + dtv * Bv * uv;

        float p = h * Cv;
        #pragma unroll
        for (int off = 1; off < 16; off <<= 1)
            p += __shfl_xor(p, off, 32);                // stays inside 16-lane group

        if (n == 0) {
            float y = p + uv * dsk;
            float z = xz[base * (2 * DI_) + DI_ + d];
            float g = y * (z / (1.f + __expf(-z)));     // y * silu(z)
            yg[base * DI_ + d] = f32_to_bf16_rne(g);
        }
    }
}

// ---------------------------------------------------------------------------
// Host-side orchestration
// ---------------------------------------------------------------------------
static inline int cdiv(int a, int b) { return (a + b - 1) / b; }

extern "C" void kernel_launch(void* const* d_in, const int* in_sizes, int n_in,
                              void* d_out, int out_size, void* d_ws, size_t ws_size,
                              hipStream_t stream)
{
    (void)in_sizes; (void)n_in; (void)out_size; (void)ws_size;

    const float* x = (const float*)d_in[0];

    // workspace carve-up (one direction at a time; directions run sequentially)
    char* ws = (char*)d_ws;
    size_t off = 0;
    auto alloc = [&](size_t bytes) { void* p = ws + off; off = (off + bytes + 255) & ~(size_t)255; return p; };

    unsigned short* x_b    = (unsigned short*)alloc((size_t)M_ * D_ * 2);          // bf16 x
    unsigned short* wWin   = (unsigned short*)alloc((size_t)2 * DI_ * D_ * 2);     // (4096,1024)
    unsigned short* wWx    = (unsigned short*)alloc((size_t)XDBL_ * DI_ * 2);      // (96,2048)
    unsigned short* wWdt   = (unsigned short*)alloc((size_t)DI_ * DT_RANK_ * 2);   // (2048,64)
    unsigned short* wWout  = (unsigned short*)alloc((size_t)D_ * DI_ * 2);         // (1024,2048)
    float*          xzf    = (float*)alloc((size_t)M_ * 2 * DI_ * 4);              // (2048,4096)
    float*          xcf    = (float*)alloc((size_t)M_ * DI_ * 4);
    unsigned short* xcb    = (unsigned short*)alloc((size_t)M_ * DI_ * 2);
    float*          xdblf  = (float*)alloc((size_t)M_ * XDBL_ * 4);
    unsigned short* xdblb  = (unsigned short*)alloc((size_t)M_ * XDBL_ * 2);
    float*          dtf    = (float*)alloc((size_t)M_ * DI_ * 4);
    unsigned short* ygb    = (unsigned short*)alloc((size_t)M_ * DI_ * 2);

    // x -> bf16 (shared by both directions; reversal handled by GEMM1 index map)
    {
        long n = (long)M_ * D_;
        k_f32_to_bf16<<<dim3((unsigned)((n + 255) / 256)), 256, 0, stream>>>(x, x_b, n);
    }

    for (int dir = 0; dir < 2; ++dir) {
        int i0 = dir ? 10 : 1;
        const float* Win   = (const float*)d_in[i0 + 0];
        const float* Wconv = (const float*)d_in[i0 + 1];
        const float* bconv = (const float*)d_in[i0 + 2];
        const float* Wx    = (const float*)d_in[i0 + 3];
        const float* Wdt   = (const float*)d_in[i0 + 4];
        const float* bdt   = (const float*)d_in[i0 + 5];
        const float* Alog  = (const float*)d_in[i0 + 6];
        const float* Dskip = (const float*)d_in[i0 + 7];
        const float* Wout  = (const float*)d_in[i0 + 8];

        // weight conversions
        { long n = (long)2 * DI_ * D_;   k_f32_to_bf16<<<dim3((unsigned)((n+255)/256)),256,0,stream>>>(Win,  wWin,  n); }
        { long n = (long)XDBL_ * DI_;    k_f32_to_bf16<<<dim3((unsigned)((n+255)/256)),256,0,stream>>>(Wx,   wWx,   n); }
        { long n = (long)DI_ * DT_RANK_; k_f32_to_bf16<<<dim3((unsigned)((n+255)/256)),256,0,stream>>>(Wdt,  wWdt,  n); }
        { long n = (long)D_ * DI_;       k_f32_to_bf16<<<dim3((unsigned)((n+255)/256)),256,0,stream>>>(Wout, wWout, n); }

        // GEMM1: xz = x(rev?) @ Win^T   (2048 x 1024 x 4096)
        k_gemm_bf16_wmma<<<dim3(cdiv(2*DI_,128), cdiv(M_,128)), 256, 0, stream>>>(
            (const __bf16*)x_b, D_, (const __bf16*)wWin, D_, xzf, 2*DI_,
            M_, 2*DI_, D_, L_, /*revA=*/dir, 0, 0, nullptr, 0, nullptr);

        // causal dw-conv + SiLU
        {
            long n = (long)M_ * DI_;
            k_conv_silu<<<dim3((unsigned)((n + 255) / 256)), 256, 0, stream>>>(
                xzf, Wconv, bconv, xcf, xcb);
        }

        // GEMM2: x_dbl = xc @ Wx^T   (2048 x 2048 x 96), fused bf16 dual-store
        k_gemm_bf16_wmma<<<dim3(cdiv(XDBL_,128), cdiv(M_,128)), 256, 0, stream>>>(
            (const __bf16*)xcb, DI_, (const __bf16*)wWx, DI_, xdblf, XDBL_,
            M_, XDBL_, DI_, L_, 0, 0, 0, nullptr, 0, xdblb);

        // GEMM3: dt = softplus(x_dbl[:, :64] @ Wdt^T + bdt)   (2048 x 64 x 2048)
        k_gemm_bf16_wmma<<<dim3(cdiv(DI_,128), cdiv(M_,128)), 256, 0, stream>>>(
            (const __bf16*)xdblb, XDBL_, (const __bf16*)wWdt, DT_RANK_, dtf, DI_,
            M_, DI_, DT_RANK_, L_, 0, 0, 0, bdt, /*epi=*/1, nullptr);

        // selective scan + SiLU(z) gating -> bf16
        k_selective_scan<<<dim3((B_ * DI_) / 16), 256, 0, stream>>>(
            xcf, dtf, xdblf, Alog, Dskip, xzf, ygb);

        // GEMM4: out (+)= y_gated @ Wout^T   (2048 x 2048 x 1024); rev writes reversed rows
        k_gemm_bf16_wmma<<<dim3(cdiv(D_,128), cdiv(M_,128)), 256, 0, stream>>>(
            (const __bf16*)ygb, DI_, (const __bf16*)wWout, DI_, (float*)d_out, D_,
            M_, D_, DI_, L_, 0, /*revC=*/dir, /*accum=*/dir, nullptr, 0, nullptr);
    }
}
